// RecurrentScaled_24739011625188
// MI455X (gfx1250) — compile-verified, matrix operand order checked by reading, not run on hardware
//
#include <hip/hip_runtime.h>
#include <hip/hip_bf16.h>

// ---------------------------------------------------------------------------
// RecurrentScaled on MI455X (gfx1250, wave32, WMMA)
//   inp = x @ W_in^T                      [32768,1024]x[1024,256]
//   20x: u = tanh((u @ W_h^T + b_h + inp) * s)
//   y   = tanh(u @ W_out^T + b_out)       [32768,256]x[256,1024]
// Fused per 128-row batch tile: inp lives in f32 accumulators, W_h fragments
// live in VGPRs, u ping-pongs through LDS as bf16, all GEMMs are
// v_wmma_f32_16x16x32_bf16. A-fragments are batch-loaded (8 per k-tile) so
// ds_load clauses overlap the WMMA bursts instead of a wait per tile.
// ---------------------------------------------------------------------------

#define BATCH 32768
#define D_IN  1024
#define HDIM  256
#define DOUT  1024
#define LAYERS 20

#define BM      128            // batch rows per workgroup
#define U_STRIDE (HDIM + 8)    // bf16 elems; 528B rows: 16B aligned, bank-spread
#define XCHUNK  64             // k-chunk of x staged in LDS per pass
#define X_STRIDE (XCHUNK + 8)  // 144B rows

typedef __attribute__((ext_vector_type(16))) __bf16       v16bf;
typedef __attribute__((ext_vector_type(8)))  float        v8f;
typedef __attribute__((ext_vector_type(4)))  unsigned int v4u;

union Frag { v16bf f; v4u u[2]; };

// Load one 16x32 bf16 WMMA A/B fragment from a row-major [rows x stride] bf16
// array. ISA layout (CDNA5 ISA, 16-bit A 16x32): lanes 0-15 hold
// rows M=0..15 with K=0..7 / K=16..23; lanes 16-31 hold K=8..15 / K=24..31.
// B (32x16, W stored row-major [N,K]) mirrors this with lane = column.
__device__ __forceinline__ v16bf load_frag(const __bf16* base, int stride,
                                           int row0, int k0, int lane) {
  const int lr   = lane & 15;
  const int half = lane >> 4;
  const __bf16* p = base + (row0 + lr) * stride + k0 + half * 8;
  Frag fr;
  fr.u[0] = *(const v4u*)(p);        // K = k0 + half*8 .. +7
  fr.u[1] = *(const v4u*)(p + 16);   // K = k0 + 16 + half*8 .. +7
  return fr.f;
}

__device__ __forceinline__ v8f bfmma(v16bf a, v16bf b, v8f c) {
  return __builtin_amdgcn_wmma_f32_16x16x32_bf16(
      /*neg_a=*/false, a, /*neg_b=*/false, b,
      /*c_mod=*/(short)0, c, /*reuse_a=*/false, /*reuse_b=*/false);
}

__device__ __forceinline__ float fast_tanh(float x) {
#if __has_builtin(__builtin_amdgcn_tanhf)
  return __builtin_amdgcn_tanhf(x);          // v_tanh_f32 (gfx1250 TRANS op)
#else
  float e = __expf(2.0f * x);                // v_exp_f32 fallback
  return 1.0f - 2.0f / (e + 1.0f);
#endif
}

// ---------------------------------------------------------------------------
// Pre-pass: convert W_in / W_h / W_out to bf16 into workspace.
// ws layout: [W_in 256*1024][W_h 256*256][W_out 1024*256]  (bf16)
// ---------------------------------------------------------------------------
#define WIN_ELEMS  (HDIM * D_IN)    // 262144
#define WH_ELEMS   (HDIM * HDIM)    // 65536
#define WOUT_ELEMS (DOUT * HDIM)    // 262144
#define WTOTAL     (WIN_ELEMS + WH_ELEMS + WOUT_ELEMS)  // 589824

__global__ void convert_weights_kernel(const float* __restrict__ Win,
                                       const float* __restrict__ Wh,
                                       const float* __restrict__ Wout,
                                       __bf16* __restrict__ dst) {
  int i = blockIdx.x * blockDim.x + threadIdx.x;
  if (i < WIN_ELEMS)                 dst[i] = (__bf16)Win[i];
  else if (i < WIN_ELEMS + WH_ELEMS) dst[i] = (__bf16)Wh[i - WIN_ELEMS];
  else if (i < WTOTAL)               dst[i] = (__bf16)Wout[i - WIN_ELEMS - WH_ELEMS];
}

// ---------------------------------------------------------------------------
// Fused recurrent kernel. Grid = BATCH/BM = 256 blocks of 256 threads (8 wave32
// waves). Wave w owns H-columns [w*32, +32) in phases 1-2 and D_OUT columns
// [w*128, +128) in phase 3. All 8 waves cover all 128 rows (8 M-tiles).
// ---------------------------------------------------------------------------
__global__ __launch_bounds__(256, 1)
void RecurrentScaled_kernel(const float* __restrict__ x,
                            const float* __restrict__ b_h,
                            const float* __restrict__ scalar,
                            const float* __restrict__ b_out,
                            const __bf16* __restrict__ Win_bf,
                            const __bf16* __restrict__ Wh_bf,
                            const __bf16* __restrict__ Wout_bf,
                            float* __restrict__ out) {
  __shared__ __align__(16) __bf16 smem[BM * U_STRIDE];   // 67.5 KB
  __bf16* lds_u = smem;          // [BM][U_STRIDE] bf16 u buffer (phases 2-3)
  __bf16* lds_x = smem;          // phase-1 x staging alias [BM][X_STRIDE]

  const int tid   = threadIdx.x;
  const int lane  = tid & 31;
  const int wave  = tid >> 5;
  const int lr    = lane & 15;
  const int half  = lane >> 4;
  const int rowbase = blockIdx.x * BM;

  const float sc = scalar[0];

  // ---------------- Phase 1: inp = x @ W_in^T ------------------------------
  v8f inp[8][2];
#pragma unroll
  for (int mt = 0; mt < 8; ++mt) {
    v8f z = {};
    inp[mt][0] = z; inp[mt][1] = z;
  }

#pragma unroll 1
  for (int kc = 0; kc < D_IN / XCHUNK; ++kc) {
    __syncthreads();   // previous chunk's readers done before overwrite
    // Cooperative coalesced stage of x[rowbase:+128, kc*64:+64] as bf16.
    for (int i = tid; i < BM * XCHUNK; i += 256) {
      int r = i >> 6;
      int k = i & (XCHUNK - 1);
      float xv = x[(rowbase + r) * D_IN + kc * XCHUNK + k];
      lds_x[r * X_STRIDE + k] = (__bf16)xv;
    }
    __syncthreads();
#pragma unroll
    for (int kt = 0; kt < 2; ++kt) {
      const int kg = kc * XCHUNK + kt * 32;
      // Batch all operand loads, then the 16-WMMA burst.
      v16bf b0 = load_frag(Win_bf, D_IN, wave * 32 + 0,  kg, lane);
      v16bf b1 = load_frag(Win_bf, D_IN, wave * 32 + 16, kg, lane);
      v16bf afrag[8];
#pragma unroll
      for (int mt = 0; mt < 8; ++mt)
        afrag[mt] = load_frag(lds_x, X_STRIDE, mt * 16, kt * 32, lane);
#pragma unroll
      for (int mt = 0; mt < 8; ++mt) {
        inp[mt][0] = bfmma(afrag[mt], b0, inp[mt][0]);
        inp[mt][1] = bfmma(afrag[mt], b1, inp[mt][1]);
      }
    }
  }

  // Fold b_h into inp once (bias is per-column; a lane holds one column).
  {
    const float bh0 = b_h[wave * 32 + lr];
    const float bh1 = b_h[wave * 32 + 16 + lr];
#pragma unroll
    for (int mt = 0; mt < 8; ++mt)
#pragma unroll
      for (int v = 0; v < 8; ++v) {
        inp[mt][0][v] += bh0;
        inp[mt][1][v] += bh1;
      }
  }

  // Preload this wave's W_h B-fragments: 2 N-tiles x 8 K-tiles = 128 VGPRs.
  v16bf whf[2][8];
#pragma unroll
  for (int nt = 0; nt < 2; ++nt)
#pragma unroll
    for (int kt = 0; kt < 8; ++kt)
      whf[nt][kt] = load_frag(Wh_bf, HDIM, wave * 32 + nt * 16, kt * 32, lane);

  // ---------------- Phase 2: 20 recurrent steps ----------------------------
#pragma unroll 1
  for (int it = 0; it < LAYERS; ++it) {
    v8f acc[8][2];
#pragma unroll
    for (int mt = 0; mt < 8; ++mt) {
      acc[mt][0] = inp[mt][0];
      acc[mt][1] = inp[mt][1];
    }

    if (it > 0) {   // first step: u==0, GEMM is a no-op
#pragma unroll
      for (int kt = 0; kt < 8; ++kt) {
        // 16 ds_load_b128 as one clause, then 16 WMMAs; scheduler can
        // overlap next k-tile's loads with this burst.
        v16bf afrag[8];
#pragma unroll
        for (int mt = 0; mt < 8; ++mt)
          afrag[mt] = load_frag(lds_u, U_STRIDE, mt * 16, kt * 32, lane);
#pragma unroll
        for (int mt = 0; mt < 8; ++mt) {
          acc[mt][0] = bfmma(afrag[mt], whf[0][kt], acc[mt][0]);
          acc[mt][1] = bfmma(afrag[mt], whf[1][kt], acc[mt][1]);
        }
      }
    }

    __syncthreads();   // all waves done reading lds_u before overwrite
    // u = tanh(acc * s): C-layout lane holds column n = wave*32+nt*16+lr,
    // rows mt*16 + v + 8*half; scatter bf16 into row-major lds_u.
#pragma unroll
    for (int mt = 0; mt < 8; ++mt)
#pragma unroll
      for (int nt = 0; nt < 2; ++nt)
#pragma unroll
        for (int v = 0; v < 8; ++v) {
          float uval = fast_tanh(acc[mt][nt][v] * sc);
          int r = mt * 16 + v + half * 8;
          int c = wave * 32 + nt * 16 + lr;
          lds_u[r * U_STRIDE + c] = (__bf16)uval;
        }
    __syncthreads();   // writes visible to all waves
  }

  // ---------------- Phase 3: y = tanh(u @ W_out^T + b_out) -----------------
#pragma unroll 1
  for (int nt = 0; nt < 8; ++nt) {
    const int colbase = wave * 128 + nt * 16;
    if (nt < 7)   // warm next W_out slice (global_prefetch_b8)
      __builtin_prefetch((const void*)(Wout_bf + (colbase + 16) * HDIM), 0, 1);
    const float bo = b_out[colbase + lr];
    v8f acc[8];
#pragma unroll
    for (int mt = 0; mt < 8; ++mt) { v8f z = {}; acc[mt] = z; }
#pragma unroll
    for (int kt = 0; kt < 8; ++kt) {
      v16bf b = load_frag(Wout_bf, HDIM, colbase, kt * 32, lane);
      v16bf afrag[8];
#pragma unroll
      for (int mt = 0; mt < 8; ++mt)
        afrag[mt] = load_frag(lds_u, U_STRIDE, mt * 16, kt * 32, lane);
#pragma unroll
      for (int mt = 0; mt < 8; ++mt)
        acc[mt] = bfmma(afrag[mt], b, acc[mt]);
    }
#pragma unroll
    for (int mt = 0; mt < 8; ++mt)
#pragma unroll
      for (int v = 0; v < 8; ++v) {
        int r = rowbase + mt * 16 + v + half * 8;
        int c = colbase + lr;
        out[r * DOUT + c] = fast_tanh(acc[mt][v] + bo);
      }
  }
}

// ---------------------------------------------------------------------------
extern "C" void kernel_launch(void* const* d_in, const int* in_sizes, int n_in,
                              void* d_out, int out_size, void* d_ws, size_t ws_size,
                              hipStream_t stream) {
  const float* x     = (const float*)d_in[0];
  const float* W_in  = (const float*)d_in[1];
  const float* W_h   = (const float*)d_in[2];
  const float* b_h   = (const float*)d_in[3];
  const float* scal  = (const float*)d_in[4];
  const float* W_out = (const float*)d_in[5];
  const float* b_out = (const float*)d_in[6];
  float* out = (float*)d_out;

  __bf16* wbf = (__bf16*)d_ws;   // needs WTOTAL*2 = 1.18 MB of workspace

  convert_weights_kernel<<<WTOTAL / 256, 256, 0, stream>>>(W_in, W_h, W_out, wbf);

  RecurrentScaled_kernel<<<BATCH / BM, 256, 0, stream>>>(
      x, b_h, scal, b_out,
      wbf,                          // W_in  bf16 [256][1024]
      wbf + WIN_ELEMS,              // W_h   bf16 [256][256]
      wbf + WIN_ELEMS + WH_ELEMS,   // W_out bf16 [1024][256]
      out);
}
